// BRD4KANModel_35811437314217
// MI455X (gfx1250) — matrix operand order repaired and checked by reading, not compiled, not run on hardware
//
#include <hip/hip_runtime.h>
#include <hip/hip_bf16.h>

// ---------- types ----------
typedef __bf16 bf16_t;
typedef __attribute__((ext_vector_type(16))) __bf16        v16bf;
typedef __attribute__((ext_vector_type(8)))  float         v8f;
typedef __attribute__((ext_vector_type(4)))  unsigned int  u32x4;

union BF16Frag { u32x4 q[2]; v16bf v; };

// Async global->LDS path (gfx1250 GLOBAL_LOAD_ASYNC_TO_LDS_B128, ASYNCcnt-tracked).
// Guarded so the host pass / older toolchains fall back to sync staging.
#if defined(__gfx1250__) && __has_builtin(__builtin_amdgcn_global_load_async_to_lds_b128)
#define USE_ASYNC_LDS 1
#else
#define USE_ASYNC_LDS 0
#endif

__device__ __forceinline__ void async_load_b128(const void* g, void* l)
{
#if USE_ASYNC_LDS
    // param types (from clang diags): 'int __vector(4) __device__ *', 'int __vector(4) __shared__ *'
    typedef int v4i_vs __attribute__((vector_size(16)));
    __builtin_amdgcn_global_load_async_to_lds_b128(
        (__attribute__((address_space(1))) v4i_vs*)g,
        (__attribute__((address_space(3))) v4i_vs*)l, 0, 0);
#else
    (void)g; (void)l;
#endif
}

__device__ __forceinline__ void wait_async_all()
{
#if USE_ASYNC_LDS
#if __has_builtin(__builtin_amdgcn_s_wait_asynccnt)
    __builtin_amdgcn_s_wait_asynccnt(0);
#else
    asm volatile("s_wait_asynccnt 0x0" ::: "memory");
#endif
#endif
}

// ---------- GEMM: C[M,N] (f32) = A[M,K] (bf16, row-major) x W[N,K]^T (bf16, row-major) ----------
// block = 256 threads = 8 waves (2 x 4). block tile 64(M) x 128(N), K-step 64,
// double-buffered LDS with async prefetch. wave tile 32x32 = 2x2 WMMA frags.
#define TM 64
#define TN 128
#define TK 64
#define STRA 72   // LDS row stride (bf16 elems), padded, multiple of 8
#define STRW 72

__device__ __forceinline__ void stage_tiles(const bf16_t* __restrict__ A,
                                            const bf16_t* __restrict__ W,
                                            bf16_t* As, bf16_t* Ws,
                                            int bm, int bn, int K, int k0, int tid)
{
    // A tile: 64 rows x 64 k = 512 chunks of 8 bf16 (16B)
#pragma unroll
    for (int it = 0; it < 2; ++it) {
        int c   = tid + it * 256;
        int row = c >> 3;
        int kc  = (c & 7) * 8;
        const bf16_t* g = A + (size_t)(bm + row) * K + k0 + kc;
        bf16_t* l = &As[row * STRA + kc];
#if USE_ASYNC_LDS
        async_load_b128(g, l);
#else
        *(u32x4*)l = *(const u32x4*)g;
#endif
    }
    // W tile: 128 rows x 64 k = 1024 chunks
#pragma unroll
    for (int it = 0; it < 4; ++it) {
        int c   = tid + it * 256;
        int row = c >> 3;
        int kc  = (c & 7) * 8;
        const bf16_t* g = W + (size_t)(bn + row) * K + k0 + kc;
        bf16_t* l = &Ws[row * STRW + kc];
#if USE_ASYNC_LDS
        async_load_b128(g, l);
#else
        *(u32x4*)l = *(const u32x4*)g;
#endif
    }
}

__global__ __launch_bounds__(256) void gemm_bf16(const bf16_t* __restrict__ A,
                                                 const bf16_t* __restrict__ W,
                                                 float* __restrict__ C,
                                                 int M, int N, int K)
{
    __shared__ alignas(16) bf16_t Asm[2][TM * STRA];
    __shared__ alignas(16) bf16_t Wsm[2][TN * STRW];
    (void)M;

    const int tid  = threadIdx.x;
    const int lane = tid & 31;
    const int wave = tid >> 5;
    const int wm   = wave >> 2;        // 0..1  (M direction)
    const int wn   = wave & 3;         // 0..3  (N direction)
    const int bm   = blockIdx.y * TM;
    const int bn   = blockIdx.x * TN;
    const int hl   = lane >> 4;        // half: 0 or 1
    const int l16  = lane & 15;

    v8f vzero = {};
    v8f acc[2][2];
#pragma unroll
    for (int i = 0; i < 2; ++i)
#pragma unroll
        for (int j = 0; j < 2; ++j) acc[i][j] = vzero;

    const int nsteps = K / TK;

    // prologue: fill buffer 0
    stage_tiles(A, W, Asm[0], Wsm[0], bm, bn, K, 0, tid);
    wait_async_all();
    __syncthreads();

    for (int s = 0; s < nsteps; ++s) {
        const int cur = s & 1;
        // async prefetch next K-step into the other buffer while we compute
        if (s + 1 < nsteps)
            stage_tiles(A, W, Asm[cur ^ 1], Wsm[cur ^ 1], bm, bn, K, (s + 1) * TK, tid);

        const bf16_t* As = Asm[cur];
        const bf16_t* Ws = Wsm[cur];
#pragma unroll
        for (int kk = 0; kk < TK; kk += 32) {
            BF16Frag a[2], b[2];
            // A 16x32 frag: lane m: elems0-7 = K(kk + hl*8 ..), elems8-15 = K(kk+16+hl*8 ..)
#pragma unroll
            for (int i = 0; i < 2; ++i) {
                int arow = wm * 32 + i * 16 + l16;
                a[i].q[0] = *(const u32x4*)&As[arow * STRA + kk + hl * 8];
                a[i].q[1] = *(const u32x4*)&As[arow * STRA + kk + 16 + hl * 8];
            }
            // B 32x16 frag: lane n holds column n (= W row), elems0-7 = K(kk + hl*8 ..), elems8-15 = K(kk+16+hl*8 ..)
#pragma unroll
            for (int j = 0; j < 2; ++j) {
                int wrow = wn * 32 + j * 16 + l16;
                b[j].q[0] = *(const u32x4*)&Ws[wrow * STRW + kk + hl * 8];
                b[j].q[1] = *(const u32x4*)&Ws[wrow * STRW + kk + 16 + hl * 8];
            }
#pragma unroll
            for (int i = 0; i < 2; ++i)
#pragma unroll
                for (int j = 0; j < 2; ++j)
                    acc[i][j] = __builtin_amdgcn_wmma_f32_16x16x32_bf16(
                        false, a[i].v, false, b[j].v, (short)0, acc[i][j], false, false);
        }
        wait_async_all();   // my prefetch landed in LDS
        __syncthreads();    // everyone's prefetch landed; everyone done reading cur
    }

    // epilogue: C/D layout: VGPR r, lanes0-15 -> M=r, lanes16-31 -> M=r+8
#pragma unroll
    for (int i = 0; i < 2; ++i) {
        int rowbase = bm + wm * 32 + i * 16 + hl * 8;
#pragma unroll
        for (int j = 0; j < 2; ++j) {
            int col = bn + wn * 32 + j * 16 + l16;
#pragma unroll
            for (int r = 0; r < 8; ++r)
                C[(size_t)(rowbase + r) * N + col] = acc[i][j][r];
        }
    }
}

// ---------- prep: h -> [silu(h) | b_splines(h)] as bf16 A-matrix row (width 7*in) ----------
// gated==1: src is hh (B x 2*in): h = sigmoid(hh[:, :in]+b[:in]) * (hh[:, in:]+b[in:])
__global__ void kan_prep(const float* __restrict__ src, const float* __restrict__ bias,
                         bf16_t* __restrict__ Acat, int B, int in, int gated)
{
    int idx = blockIdx.x * blockDim.x + threadIdx.x;
    if (idx >= B * in) return;
    int b = idx / in, i = idx - b * in;

    float h;
    if (gated) {
        float gte = src[(size_t)b * 2 * in + i]      + bias[i];
        float val = src[(size_t)b * 2 * in + in + i] + bias[in + i];
        h = val / (1.f + __expf(-gte));
    } else {
        h = src[idx];
    }

    size_t rb = (size_t)b * (size_t)(7 * in);
    // base path: silu(h)
    Acat[rb + i] = (bf16_t)(h / (1.f + __expf(-h)));

    // Cox-de-Boor, grid_size=3, order=3: g_j = (j-3)*(2/3) - 1, j=0..9
    const float hstep = 2.0f / 3.0f;
    float g[10];
#pragma unroll
    for (int j = 0; j < 10; ++j) g[j] = (float)(j - 3) * hstep - 1.0f;
    float bs[9];
#pragma unroll
    for (int j = 0; j < 9; ++j) bs[j] = (h >= g[j] && h < g[j + 1]) ? 1.f : 0.f;
#pragma unroll
    for (int k = 1; k <= 3; ++k) {
#pragma unroll
        for (int j = 0; j < 9 - k; ++j) {
            bs[j] = (h - g[j]) / (g[j + k] - g[j]) * bs[j]
                  + (g[j + k + 1] - h) / (g[j + k + 1] - g[j + 1]) * bs[j + 1];
        }
    }
#pragma unroll
    for (int c = 0; c < 6; ++c)
        Acat[rb + in + (size_t)i * 6 + c] = (bf16_t)bs[c];
}

// ---------- weight concat: Wcat[o, :] = [ base_w[o,:] | spline_w[o,i,c]*scaler[o,i] ] -> bf16 ----------
__global__ void build_wcat(const float* __restrict__ bw, const float* __restrict__ sw,
                           const float* __restrict__ sc, bf16_t* __restrict__ Wc,
                           int out, int in)
{
    size_t total = (size_t)out * 7 * in;
    size_t idx = (size_t)blockIdx.x * blockDim.x + threadIdx.x;
    if (idx >= total) return;
    int kcw = 7 * in;
    int row = (int)(idx / kcw);
    int r   = (int)(idx - (size_t)row * kcw);
    float v;
    if (r < in) {
        v = bw[(size_t)row * in + r];
    } else {
        int j = r - in;
        int i = j / 6, c = j - i * 6;
        v = sw[((size_t)row * in + i) * 6 + c] * sc[(size_t)row * in + i];
    }
    Wc[idx] = (bf16_t)v;
}

// ---------- fp32 -> bf16 ----------
__global__ void cvt_bf16(const float* __restrict__ s, bf16_t* __restrict__ d, size_t n)
{
    size_t i = (size_t)blockIdx.x * blockDim.x + threadIdx.x;
    if (i < n) d[i] = (bf16_t)s[i];
}

// ---------- reg/aux heads: one wave per batch row ----------
__global__ __launch_bounds__(256) void heads(const float* __restrict__ h3,
                                             const float* __restrict__ rw, const float* __restrict__ rb,
                                             const float* __restrict__ aw, const float* __restrict__ ab,
                                             float* __restrict__ out, int B, int n)
{
    int wave = threadIdx.x >> 5, lane = threadIdx.x & 31;
    int row = blockIdx.x * 8 + wave;
    if (row >= B) return;
    float sr = 0.f, sa = 0.f;
    for (int j = lane; j < n; j += 32) {
        float v = h3[(size_t)row * n + j];
        sr += v * rw[j];
        sa += v * aw[j];
    }
#pragma unroll
    for (int off = 16; off; off >>= 1) {
        sr += __shfl_down(sr, off, 32);
        sa += __shfl_down(sa, off, 32);
    }
    if (lane == 0) {
        out[row]     = sr + rb[0];
        out[B + row] = sa + ab[0];
    }
}

// ---------- launch ----------
extern "C" void kernel_launch(void* const* d_in, const int* in_sizes, int n_in,
                              void* d_out, int out_size, void* d_ws, size_t ws_size,
                              hipStream_t stream)
{
    (void)in_sizes; (void)n_in; (void)out_size; (void)ws_size;
    const float* x   = (const float*)d_in[0];
    const float* mw  = (const float*)d_in[1];
    const float* mb  = (const float*)d_in[2];
    const float* bw0 = (const float*)d_in[3];
    const float* sw0 = (const float*)d_in[4];
    const float* sc0 = (const float*)d_in[5];
    const float* bw1 = (const float*)d_in[6];
    const float* sw1 = (const float*)d_in[7];
    const float* sc1 = (const float*)d_in[8];
    const float* bw2 = (const float*)d_in[9];
    const float* sw2 = (const float*)d_in[10];
    const float* sc2 = (const float*)d_in[11];
    const float* rw  = (const float*)d_in[12];
    const float* rb  = (const float*)d_in[13];
    const float* aw  = (const float*)d_in[14];
    const float* ab  = (const float*)d_in[15];
    float* out = (float*)d_out;

    constexpr int B = 4096, D = 2048;
    constexpr int KC = 7 * D;       // 14336 = 2048 base + 2048*6 spline
    constexpr int O2 = 1024;

    char* w = (char*)d_ws;
    size_t off = 0;
    auto alloc = [&](size_t bytes) -> void* {
        void* p = (void*)(w + off);
        off += (bytes + 255) & ~(size_t)255;
        return p;
    };
    bf16_t* xb  = (bf16_t*)alloc((size_t)B * D * 2);          // x in bf16
    bf16_t* mwb = (bf16_t*)alloc((size_t)2 * D * D * 2);      // mult_w in bf16
    bf16_t* wc0 = (bf16_t*)alloc((size_t)D  * KC * 2);        // [base|spline] layer0
    bf16_t* wc1 = (bf16_t*)alloc((size_t)D  * KC * 2);
    bf16_t* wc2 = (bf16_t*)alloc((size_t)O2 * KC * 2);
    float*  hh  = (float*)alloc((size_t)B * 2 * D * 4);       // gate|val pre-activation
    bf16_t* ac  = (bf16_t*)alloc((size_t)B * KC * 2);         // A-matrix [silu|bases]
    float*  h3  = (float*)alloc((size_t)B * O2 * 4);
    float*  h1  = hh;                                         // overlay: hh dead after prep0
    float*  h2  = hh + (size_t)B * D;

    // weight / input conversion (bandwidth-bound, ~µs at 23.3 TB/s)
    cvt_bf16<<<(B * D + 255) / 256, 256, 0, stream>>>(x, xb, (size_t)B * D);
    cvt_bf16<<<(2 * D * D + 255) / 256, 256, 0, stream>>>(mw, mwb, (size_t)2 * D * D);
    build_wcat<<<(int)(((size_t)D  * KC + 255) / 256), 256, 0, stream>>>(bw0, sw0, sc0, wc0, D,  D);
    build_wcat<<<(int)(((size_t)D  * KC + 255) / 256), 256, 0, stream>>>(bw1, sw1, sc1, wc1, D,  D);
    build_wcat<<<(int)(((size_t)O2 * KC + 255) / 256), 256, 0, stream>>>(bw2, sw2, sc2, wc2, O2, D);

    // hh = x @ mult_w^T        (M=4096, N=4096, K=2048)
    gemm_bf16<<<dim3(2 * D / TN, B / TM), 256, 0, stream>>>(xb, mwb, hh, B, 2 * D, D);
    // h = sigmoid(gate)*val -> A0 = [silu(h) | bases(h)]
    kan_prep<<<(B * D + 255) / 256, 256, 0, stream>>>(hh, mb, ac, B, D, 1);
    // layer 0 (fused base+spline GEMM, K=14336)
    gemm_bf16<<<dim3(D / TN, B / TM), 256, 0, stream>>>(ac, wc0, h1, B, D, KC);
    kan_prep<<<(B * D + 255) / 256, 256, 0, stream>>>(h1, mb, ac, B, D, 0);
    // layer 1
    gemm_bf16<<<dim3(D / TN, B / TM), 256, 0, stream>>>(ac, wc1, h2, B, D, KC);
    kan_prep<<<(B * D + 255) / 256, 256, 0, stream>>>(h2, mb, ac, B, D, 0);
    // layer 2 (out = 1024)
    gemm_bf16<<<dim3(O2 / TN, B / TM), 256, 0, stream>>>(ac, wc2, h3, B, O2, KC);
    // reg / aux heads
    heads<<<B / 8, 256, 0, stream>>>(h3, rw, rb, aw, ab, out, B, O2);
}
